// HistoryEncoder_78288663872280
// MI455X (gfx1250) — matrix operand order repaired and checked by reading, not compiled, not run on hardware
//
#include <hip/hip_runtime.h>
#include <hip/hip_bf16.h>

// ---------------------------------------------------------------------------
// HistoryEncoder for MI455X (gfx1250, wave32, WMMA).
// Pipeline: prep (weight->WMMA-B-frag f16) ; alpha (gate MLP) ;
//           lstm (15-step recurrence, v_wmma_f32_16x16x32_f16) ;
//           proj_fuse_ln (h_n @ W_rel^T WMMA + fuse + LayerNorm).
// ---------------------------------------------------------------------------

typedef __attribute__((ext_vector_type(16))) _Float16 v16h;
typedef __attribute__((ext_vector_type(8)))  _Float16 v8h;
typedef __attribute__((ext_vector_type(8)))  float    v8f;

#define D_DIM  1024
#define HID    512
#define GDIM   2048        // 4*HID
#define HLEN   15
#define BN_TOT 16384       // 32*512

// workspace layout (bytes)
#define ALPHA_OFF 0
#define WHH_OFF   1024
#define WREL_OFF  (WHH_OFF + GDIM*HID*2)          // +2MB
#define HN_OFF    (WREL_OFF + D_DIM*HID*2)        // +1MB
#define CW_OFF    (HN_OFF + (size_t)BN_TOT*HID*2) // +16MB ; c-state 32MB f32

__device__ __forceinline__ float sigmoidf_(float x) {
    return 1.0f / (1.0f + __expf(-x));
}

// ---------------------------------------------------------------------------
// Prep: build pre-swizzled WMMA B-fragments (f16).
// Fragment layout per 32(K) x 16(N) tile: lane l holds 16 halfs contiguous:
//   element j of lane l  ->  K = kb*32 + (l/16)*16 + j , N = nb*16 + (l%16)
// contiguous K order per lane half, the same order the A operand loads use (two b128 loads per operand).
// whhF: B[k][g] = W_hh[g][k]  (col tiles: 128 = 4 gate blocks * 32 hid tiles)
// wrelF: B[k][d] = W_rel[d][k] (col tiles: 64)
// ---------------------------------------------------------------------------
__global__ void prep_kernel(const float* __restrict__ Whh,
                            const float* __restrict__ Wrel,
                            _Float16* __restrict__ whhF,
                            _Float16* __restrict__ wrelF) {
    const int NWHH  = GDIM * HID;
    const int NWREL = D_DIM * HID;
    int i = blockIdx.x * blockDim.x + threadIdx.x;
    if (i >= NWHH + NWREL) return;
    if (i < NWHH) {
        int j = i & 15, lane = (i >> 4) & 31, tile = i >> 9;
        int cb = tile & 127, kb = tile >> 7;
        int g = cb * 16 + (lane & 15);
        int k = kb * 32 + ((lane >> 4) << 4) + j;
        whhF[i] = (_Float16)Whh[g * HID + k];
    } else {
        int i2 = i - NWHH;
        int j = i2 & 15, lane = (i2 >> 4) & 31, tile = i2 >> 9;
        int nb = tile & 63, kb = tile >> 6;
        int n = nb * 16 + (lane & 15);
        int k = kb * 32 + ((lane >> 4) << 4) + j;
        wrelF[i2] = (_Float16)Wrel[n * HID + k];
    }
}

// ---------------------------------------------------------------------------
// Gate MLP: alpha[b] = sigmoid(W_g2 @ relu(W_g1 @ [ego,gps] + b_g1) + b_g2)
// ---------------------------------------------------------------------------
__global__ void alpha_kernel(const float* __restrict__ ego,
                             const float* __restrict__ gps,
                             const float* __restrict__ Wg1,
                             const float* __restrict__ bg1,
                             const float* __restrict__ Wg2,
                             const float* __restrict__ bg2,
                             float* __restrict__ alpha) {
    int b = threadIdx.x;
    if (b < 32) {
        float c0 = ego[b], c1 = gps[b];
        float acc = bg2[0];
#pragma unroll
        for (int j = 0; j < 16; ++j) {
            float h = Wg1[2 * j] * c0 + Wg1[2 * j + 1] * c1 + bg1[j];
            h = h > 0.f ? h : 0.f;
            acc += Wg2[j] * h;
        }
        alpha[b] = sigmoidf_(acc);
    }
}

// ---------------------------------------------------------------------------
// LSTM: 256 blocks x 64 rows. Per step, per wave: 4 hidden tiles * 4 gates
// * 4 row tiles * 16 K-steps of v_wmma_f32_16x16x32_f16. The D-tile lane
// layout (lane l -> col l%16, rows r+8*(l/16)) puts i/f/g/o for one
// (row, hidden) pair in the SAME lane -> pure per-lane gate math.
// h double-buffered in LDS (f16); c streamed f32 through workspace.
// ---------------------------------------------------------------------------
__global__ __launch_bounds__(256, 1)
void lstm_kernel(const float* __restrict__ traj,
                 const float* __restrict__ Wih,
                 const float* __restrict__ b_ih,
                 const float* __restrict__ b_hh,
                 const _Float16* __restrict__ whhF,
                 float* __restrict__ cws,
                 _Float16* __restrict__ hn) {
    __shared__ _Float16 hbuf[2][64 * HID];   // 2 x 64KB

    const int tid = threadIdx.x;
    const int wv = tid >> 5;
    const int lane = tid & 31;
    const int bn0 = blockIdx.x * 64;

    // zero h(t=0)
    {
        uint32_t* p = (uint32_t*)&hbuf[0][0];
        for (int i = tid; i < 64 * HID / 2; i += 256) p[i] = 0u;
    }

    for (int t = 0; t < HLEN; ++t) {
        __syncthreads();
        const int cur = t & 1, nxt = cur ^ 1;
#pragma unroll 1
        for (int q = 0; q < 4; ++q) {
            const int hb = wv * 4 + q;            // hidden col tile 0..31
            v8f acc[4][4];                        // [gate][row tile]
#pragma unroll
            for (int g = 0; g < 4; ++g)
#pragma unroll
                for (int rt = 0; rt < 4; ++rt) { v8f z = {}; acc[g][rt] = z; }

#pragma unroll 1
            for (int kb = 0; kb < 16; ++kb) {
                v16h bfr[4];
#pragma unroll
                for (int g = 0; g < 4; ++g) {
                    const v16h* bp = (const v16h*)(whhF +
                        (size_t)(kb * 128 + g * 32 + hb) * 512 + lane * 16);
                    bfr[g] = *bp;                 // 2x global_load_b128
                }
                // prefetch next kb's B fragments into cache (global_prefetch_b8)
                if (kb < 15) {
#pragma unroll
                    for (int g = 0; g < 4; ++g) {
                        const _Float16* pp = whhF +
                            (size_t)((kb + 1) * 128 + g * 32 + hb) * 512 + lane * 16;
                        __builtin_prefetch((const void*)pp, 0, 3);
                    }
                }
#pragma unroll
                for (int rt = 0; rt < 4; ++rt) {
                    // A operand: lane l holds row l%16; K chunks split by lane half
                    const _Float16* hr = &hbuf[cur][(rt * 16 + (lane & 15)) * HID
                                                    + kb * 32 + ((lane >> 4) << 4)];
                    union { v16h v; v8h h[2]; } ua;
                    ua.h[0] = *(const v8h*)hr;        // K = base..base+7
                    ua.h[1] = *(const v8h*)(hr + 8);  // K = base+8..base+15
#pragma unroll
                    for (int g = 0; g < 4; ++g)
                        acc[g][rt] = __builtin_amdgcn_wmma_f32_16x16x32_f16(
                            false, ua.v, false, bfr[g], (short)0, acc[g][rt],
                            false, false);
                }
            }

            // ---- elementwise gate update (per-lane: all 4 gates in-lane) ----
            const int hcol = hb * 16 + (lane & 15);
            float wih[4][4], bsum[4];
#pragma unroll
            for (int g = 0; g < 4; ++g) {
                int gc = g * HID + hcol;
                bsum[g] = b_ih[gc] + b_hh[gc];
#pragma unroll
                for (int ii = 0; ii < 4; ++ii) wih[g][ii] = Wih[gc * 4 + ii];
            }
#pragma unroll
            for (int rt = 0; rt < 4; ++rt) {
#pragma unroll
                for (int r = 0; r < 8; ++r) {
                    const int rl = rt * 16 + r + ((lane >> 4) << 3);
                    const int bn = bn0 + rl;
                    const float* xp = traj + ((size_t)bn * HLEN + t) * 4;
                    const float x0 = xp[0], x1 = xp[1], x2 = xp[2], x3 = xp[3];
                    float pre[4];
#pragma unroll
                    for (int g = 0; g < 4; ++g)
                        pre[g] = acc[g][rt][r] + bsum[g] + wih[g][0] * x0 +
                                 wih[g][1] * x1 + wih[g][2] * x2 + wih[g][3] * x3;
                    const float iv = sigmoidf_(pre[0]);
                    const float fv = sigmoidf_(pre[1]);
                    const float gv = tanhf(pre[2]);
                    const float ov = sigmoidf_(pre[3]);
                    const size_t ci = (size_t)bn * HID + hcol;
                    const float c_old = (t == 0) ? 0.f : cws[ci];
                    const float cn = fv * c_old + iv * gv;
                    cws[ci] = cn;
                    const float hv = ov * tanhf(cn);
                    hbuf[nxt][rl * HID + hcol] = (_Float16)hv;
                    if (t == HLEN - 1) hn[ci] = (_Float16)hv;
                }
            }
        }
    }
}

// ---------------------------------------------------------------------------
// Projection + fuse + LayerNorm. 1024 blocks x 16 rows.
// h_rel tile via WMMA; fused x staged in 64KB LDS; LN with shfl_xor(16).
// ---------------------------------------------------------------------------
__global__ __launch_bounds__(256, 2)
void proj_fuse_ln_kernel(const float* __restrict__ tokens,
                         const float* __restrict__ habs,
                         const float* __restrict__ Wabs,
                         const float* __restrict__ gamma,
                         const float* __restrict__ beta,
                         const float* __restrict__ alphaArr,
                         const _Float16* __restrict__ hn,
                         const _Float16* __restrict__ wrelF,
                         float* __restrict__ out) {
    __shared__ float xbuf[16 * D_DIM];  // 64KB

    const int tid = threadIdx.x;
    const int wv = tid >> 5;
    const int lane = tid & 31;
    const int bn0 = blockIdx.x * 16;
    const float alpha = alphaArr[blockIdx.x >> 5];   // 32 blocks per batch

#pragma unroll 1
    for (int c8 = 0; c8 < 8; ++c8) {
        const int cb = wv * 8 + c8;                  // col tile 0..63
        v8f acc = {};
#pragma unroll 1
        for (int kb = 0; kb < 16; ++kb) {
            const _Float16* hr = hn + (size_t)(bn0 + (lane & 15)) * HID
                                 + kb * 32 + ((lane >> 4) << 4);
            union { v16h v; v8h h[2]; } ua;
            ua.h[0] = *(const v8h*)hr;
            ua.h[1] = *(const v8h*)(hr + 8);
            const v16h* bp = (const v16h*)(wrelF +
                (size_t)(kb * 64 + cb) * 512 + lane * 16);
            if (kb < 15) {
                __builtin_prefetch((const void*)(wrelF +
                    (size_t)((kb + 1) * 64 + cb) * 512 + lane * 16), 0, 3);
            }
            acc = __builtin_amdgcn_wmma_f32_16x16x32_f16(
                false, ua.v, false, *bp, (short)0, acc, false, false);
        }
        const int d = cb * 16 + (lane & 15);
        const float wa0 = Wabs[d * 2], wa1 = Wabs[d * 2 + 1];
#pragma unroll
        for (int r = 0; r < 8; ++r) {
            const int row = r + ((lane >> 4) << 3);
            const int bn = bn0 + row;
            const float hv = habs[bn * 2] * wa0 + habs[bn * 2 + 1] * wa1;
            const float xv = tokens[(size_t)bn * D_DIM + d] +
                             alpha * acc[r] + (1.f - alpha) * hv;
            xbuf[row * D_DIM + d] = xv;
        }
    }
    __syncthreads();

    // LayerNorm: 16 threads per row
    const int row = tid >> 4;
    const int j0 = tid & 15;
    float s = 0.f, s2 = 0.f;
#pragma unroll 1
    for (int k = 0; k < 64; ++k) {
        float v = xbuf[row * D_DIM + j0 + k * 16];
        s += v; s2 += v * v;
    }
#pragma unroll
    for (int off = 8; off >= 1; off >>= 1) {
        s  += __shfl_xor(s,  off, 16);
        s2 += __shfl_xor(s2, off, 16);
    }
    const float mu = s * (1.0f / 1024.0f);
    const float var = s2 * (1.0f / 1024.0f) - mu * mu;
    const float inv = rsqrtf(var + 1e-5f);
    const size_t ro = (size_t)(bn0 + row) * D_DIM;
#pragma unroll 1
    for (int k = 0; k < 64; ++k) {
        int d = j0 + k * 16;
        float v = xbuf[row * D_DIM + d];
        out[ro + d] = (v - mu) * inv * gamma[d] + beta[d];
    }
}

// ---------------------------------------------------------------------------
extern "C" void kernel_launch(void* const* d_in, const int* in_sizes, int n_in,
                              void* d_out, int out_size, void* d_ws, size_t ws_size,
                              hipStream_t stream) {
    const float* tokens = (const float*)d_in[0];
    const float* traj   = (const float*)d_in[1];
    const float* habs   = (const float*)d_in[2];
    const float* ego    = (const float*)d_in[3];
    const float* gps    = (const float*)d_in[4];
    const float* Wih    = (const float*)d_in[5];
    const float* Whh    = (const float*)d_in[6];
    const float* bih    = (const float*)d_in[7];
    const float* bhh    = (const float*)d_in[8];
    const float* Wrel   = (const float*)d_in[9];
    const float* Wabs   = (const float*)d_in[10];
    const float* Wg1    = (const float*)d_in[11];
    const float* bg1    = (const float*)d_in[12];
    const float* Wg2    = (const float*)d_in[13];
    const float* bg2    = (const float*)d_in[14];
    const float* gamma  = (const float*)d_in[15];
    const float* beta   = (const float*)d_in[16];

    char* wsb = (char*)d_ws;
    float*    alpha = (float*)(wsb + ALPHA_OFF);
    _Float16* whhF  = (_Float16*)(wsb + WHH_OFF);
    _Float16* wrelF = (_Float16*)(wsb + WREL_OFF);
    _Float16* hn    = (_Float16*)(wsb + HN_OFF);
    float*    cws   = (float*)(wsb + CW_OFF);
    float*    out   = (float*)d_out;

    const int nprep = GDIM * HID + D_DIM * HID;   // 1,572,864 elements
    prep_kernel<<<(nprep + 255) / 256, 256, 0, stream>>>(Whh, Wrel, whhF, wrelF);
    alpha_kernel<<<1, 32, 0, stream>>>(ego, gps, Wg1, bg1, Wg2, bg2, alpha);
    lstm_kernel<<<BN_TOT / 64, 256, 0, stream>>>(traj, Wih, bih, bhh, whhF, cws, hn);
    proj_fuse_ln_kernel<<<BN_TOT / 16, 256, 0, stream>>>(
        tokens, habs, Wabs, gamma, beta, alpha, hn, wrelF, out);
}
